// Block_7713761264306
// MI455X (gfx1250) — matrix operand order repaired and checked by reading, not compiled, not run on hardware
//
#include <hip/hip_runtime.h>
#include <hip/hip_bf16.h>
#include <math.h>

// ---------------------------------------------------------------------------
// Types
// ---------------------------------------------------------------------------
typedef __attribute__((ext_vector_type(16))) __bf16       v16bf;
typedef __attribute__((ext_vector_type(8)))  float        v8f;
typedef __attribute__((ext_vector_type(4)))  unsigned int u32x4;
typedef __attribute__((ext_vector_type(2)))  unsigned int u32x2;
typedef __attribute__((ext_vector_type(4)))  float        f32x4;

union FragAB {
  v16bf v;
  u32x4 q[2];
  unsigned short u[16];
};

__device__ __forceinline__ unsigned short f2bf(float f) {
  union { float f; unsigned int u; } c; c.f = f;
  unsigned int r = (c.u + 0x7FFFu + ((c.u >> 16) & 1u)) >> 16;
  return (unsigned short)r;
}

__device__ __forceinline__ float gelu_exact(float x) {
  return 0.5f * x * (1.0f + erff(x * 0.70710678118654752f));
}

#define WMMA_BF16(A, B, C) \
  __builtin_amdgcn_wmma_f32_16x16x32_bf16(false, (A), false, (B), (short)0, (C), false, false)

// v_permlane16_b32 with a static xor pattern (VALU cross-lane, row of 16).
template<unsigned S0, unsigned S1>
__device__ __forceinline__ float pl16(float x) {
  union { float f; unsigned u; } c; c.f = x;
  c.u = __builtin_amdgcn_permlane16(c.u, c.u, S0, S1, false, false);
  return c.f;
}
// max-reduce across the 16 lanes of a half-row (butterfly xor 1,2,4,8)
__device__ __forceinline__ float rowmax16(float x) {
  x = fmaxf(x, pl16<0x67452301u, 0xEFCDAB89u>(x));  // xor 1
  x = fmaxf(x, pl16<0x54761032u, 0xDCFE98BAu>(x));  // xor 2
  x = fmaxf(x, pl16<0x32107654u, 0xBA98FEDCu>(x));  // xor 4
  x = fmaxf(x, pl16<0xFEDCBA98u, 0x76543210u>(x));  // xor 8
  return x;
}

// ---------------------------------------------------------------------------
// fp32 -> bf16 conversion (weights)
// ---------------------------------------------------------------------------
__global__ void f2bf_kernel(const float* __restrict__ in,
                            unsigned short* __restrict__ out, int n) {
  int i = blockIdx.x * blockDim.x + threadIdx.x;
  int stride = gridDim.x * blockDim.x;
  for (; i < n; i += stride) out[i] = f2bf(in[i]);
}

// ---------------------------------------------------------------------------
// LayerNorm (D = 1024 fixed), one 256-thread block per row, bf16 output.
// ---------------------------------------------------------------------------
template<bool DO_GELU>
__global__ void layernorm_kernel(const float* __restrict__ x,
                                 const float* __restrict__ g,
                                 const float* __restrict__ b,
                                 unsigned short* __restrict__ y) {
  const int D = 1024;
  int row = blockIdx.x;
  int tid = threadIdx.x;
  const float* xr = x + (size_t)row * D;
  f32x4 xv = *(const f32x4*)(xr + tid * 4);
  float s1 = xv.x + xv.y + xv.z + xv.w;
  float s2 = xv.x * xv.x + xv.y * xv.y + xv.z * xv.z + xv.w * xv.w;
  #pragma unroll
  for (int m = 16; m >= 1; m >>= 1) {
    s1 += __shfl_xor(s1, m);
    s2 += __shfl_xor(s2, m);
  }
  __shared__ float sh1[8], sh2[8];
  if ((tid & 31) == 0) { sh1[tid >> 5] = s1; sh2[tid >> 5] = s2; }
  __syncthreads();
  float t1 = 0.f, t2 = 0.f;
  #pragma unroll
  for (int i = 0; i < 8; i++) { t1 += sh1[i]; t2 += sh2[i]; }
  float mu  = t1 * (1.0f / D);
  float var = t2 * (1.0f / D) - mu * mu;
  float rs  = rsqrtf(var + 1e-5f);
  f32x4 gv = *(const f32x4*)(g + tid * 4);
  f32x4 bv = *(const f32x4*)(b + tid * 4);
  union { unsigned short s[4]; u32x2 d; } p;
  #pragma unroll
  for (int i = 0; i < 4; i++) {
    float v = (xv[i] - mu) * rs * gv[i] + bv[i];
    if (DO_GELU) v = gelu_exact(v);
    p.s[i] = f2bf(v);
  }
  *(u32x2*)(y + (size_t)row * D + tid * 4) = p.d;
}

// ---------------------------------------------------------------------------
// K-transpose: qkv[:,1024:2048] -> kT[B*H][64][T] (feature-major per head).
// Fill phase uses CDNA5 async global->LDS copies (ASYNCcnt) instead of a
// VGPR round-trip; drain with s_wait_asynccnt before the barrier.
// ---------------------------------------------------------------------------
__global__ void ktranspose_kernel(const unsigned short* __restrict__ qkv,
                                  unsigned short* __restrict__ kT) {
  const int T = 2048, HSTR = 3072;
  __shared__ __align__(16) unsigned short tile[64][72];
  int blk  = blockIdx.x;        // B*H*(T/64) = 2048
  int slab = blk & 31;          // T/64
  int bh   = blk >> 5;          // 0..63
  int b = bh >> 4, h = bh & 15;
  int t0 = slab * 64;
  int tid = threadIdx.x;
  {
    int tok = tid >> 2;               // 0..63
    int f0  = (tid & 3) * 16;         // 0,16,32,48
    const unsigned short* src =
        qkv + (size_t)(b * T + t0 + tok) * HSTR + 1024 + h * 64 + f0;
    // LDS byte offsets relative to the (sole) shared allocation.
    unsigned lds0 = (unsigned)((char*)&tile[tok][f0] - (char*)&tile[0][0]);
    unsigned lds1 = lds0 + 16u;
    asm volatile("global_load_async_to_lds_b128 %0, %1, off"
                 :: "v"(lds0), "v"(src) : "memory");
    asm volatile("global_load_async_to_lds_b128 %0, %1, off"
                 :: "v"(lds1), "v"(src + 8) : "memory");
    asm volatile("s_wait_asynccnt 0" ::: "memory");
  }
  __syncthreads();
  {
    int f  = tid >> 2;                // 0..63
    int tc = (tid & 3) * 16;          // token chunk
    union { unsigned short s[16]; u32x4 q[2]; } p;
    #pragma unroll
    for (int i = 0; i < 16; i++) p.s[i] = tile[tc + i][f];
    unsigned short* dst = kT + ((size_t)bh * 64 + f) * T + t0 + tc;
    *(u32x4*)dst       = p.q[0];
    *(u32x4*)(dst + 8) = p.q[1];
  }
}

// ---------------------------------------------------------------------------
// GEMM: C[M,N] = epilogue(A[M,K] @ W[K,N] + bias), A/W bf16, fp32 accumulate.
// Each wave computes a 32x64 tile: 2 A-fragments x 4 B-fragments = 8 WMMAs
// per 32-deep K step; B fragments are reused across both M halves.
// ---------------------------------------------------------------------------
template<int ACT, bool HAS_RES, bool OUT_F, bool OUT_B>
__global__ void gemm_bf16_wmma(const unsigned short* __restrict__ A,
                               const unsigned short* __restrict__ W,
                               const float* __restrict__ bias,
                               const float* __restrict__ res,
                               float* __restrict__ outF,
                               unsigned short* __restrict__ outB,
                               int M, int N, int K) {
  int wave = threadIdx.x >> 5;
  int lane = threadIdx.x & 31;
  int hl = lane >> 4;
  int r  = lane & 15;
  int tilesN = N >> 6;
  int totalTiles = (M >> 5) * tilesN;
  int t = blockIdx.x * 8 + wave;
  if (t >= totalTiles) return;
  int mTile = t / tilesN;
  int nTile = t - mTile * tilesN;

  v8f acc[2][4] = {{{}, {}, {}, {}}, {{}, {}, {}, {}}};
  const unsigned short* arow0 = A + (size_t)(mTile * 32 + r) * K;
  const unsigned short* arow1 = arow0 + (size_t)16 * K;
  const unsigned short* wbase = W + nTile * 64 + r;

  for (int k0 = 0; k0 < K; k0 += 32) {
    FragAB a0, a1;
    a0.q[0] = *(const u32x4*)(arow0 + k0 + hl * 8);
    a0.q[1] = *(const u32x4*)(arow0 + k0 + 16 + hl * 8);
    a1.q[0] = *(const u32x4*)(arow1 + k0 + hl * 8);
    a1.q[1] = *(const u32x4*)(arow1 + k0 + 16 + hl * 8);
    const unsigned short* wrow = wbase + (size_t)k0 * N;
    if (k0 + 64 < K) __builtin_prefetch(wbase + (size_t)(k0 + 64) * N, 0, 1);
    #pragma unroll
    for (int j = 0; j < 4; j++) {
      FragAB bfr;
      #pragma unroll
      for (int s = 0; s < 16; s++)
        bfr.u[s] = wrow[(size_t)((s & 7) + hl * 8 + (s >> 3) * 16) * N + j * 16];
      acc[0][j] = WMMA_BF16(a0.v, bfr.v, acc[0][j]);
      acc[1][j] = WMMA_BF16(a1.v, bfr.v, acc[1][j]);
    }
  }

  int colBase = nTile * 64 + r;
  #pragma unroll
  for (int i = 0; i < 2; i++) {
    int rowBase = mTile * 32 + i * 16 + hl * 8;
    #pragma unroll
    for (int j = 0; j < 4; j++) {
      int col = colBase + j * 16;
      float bs = bias[col];
      #pragma unroll
      for (int v = 0; v < 8; v++) {
        int row = rowBase + v;
        float val = acc[i][j][v] + bs;
        if (ACT == 1) val = gelu_exact(val);
        size_t idx = (size_t)row * N + col;
        if (HAS_RES) val += res[idx];
        if (OUT_F) outF[idx] = val;
        if (OUT_B) outB[idx] = f2bf(val);
      }
    }
  }
}

// ---------------------------------------------------------------------------
// Flash attention (no mask). qkv: [B*T, 3072] bf16, kT: [B*H][64][T] bf16.
// One wave per (b, h, 16-query tile); 32-key tiles, online softmax.
// Row max: v_permlane16 butterflies (VALU only). Row sum: an extra WMMA
// against an all-ones B fragment (Lacc accumulator, rescaled like O).
// ---------------------------------------------------------------------------
__global__ void attention_kernel(const unsigned short* __restrict__ qkv,
                                 const unsigned short* __restrict__ kT,
                                 unsigned short* __restrict__ o) {
  const int T = 2048, HSTR = 3072, HS = 64;
  __shared__ __align__(16) unsigned short Pst[8][16][32];  // per-wave P tile
  int wave = threadIdx.x >> 5;
  int lane = threadIdx.x & 31;
  int hl = lane >> 4;
  int r  = lane & 15;
  int wg = blockIdx.x * 8 + wave;   // 0 .. 8191
  int bh = wg >> 7;                 // 128 query tiles per (b,h)
  int tq = wg & 127;
  int b = bh >> 4;
  int h = bh & 15;
  int t0 = tq * 16;

  size_t base = (size_t)b * T * HSTR;
  const unsigned short* qhead  = qkv + base + (size_t)t0 * HSTR + h * HS;
  const unsigned short* vhead  = qkv + base + 2048 + h * HS;
  const unsigned short* kTbase = kT + (size_t)bh * 64 * T;

  FragAB qa0, qa1;
  {
    const unsigned short* qr = qhead + (size_t)r * HSTR;
    qa0.q[0] = *(const u32x4*)(qr + hl * 8);
    qa0.q[1] = *(const u32x4*)(qr + 16 + hl * 8);
    qa1.q[0] = *(const u32x4*)(qr + 32 + hl * 8);
    qa1.q[1] = *(const u32x4*)(qr + 48 + hl * 8);
  }

  FragAB ones;
  #pragma unroll
  for (int i = 0; i < 16; i++) ones.u[i] = 0x3F80;

  float mrow[8];
  v8f O0 = {}, O1 = {}, O2 = {}, O3 = {}, Lacc = {};
  #pragma unroll
  for (int v = 0; v < 8; v++) mrow[v] = -INFINITY;

  for (int kt0 = 0; kt0 < T; kt0 += 32) {
    const unsigned short* k0p = qkv + base + (size_t)(kt0 + r) * HSTR + 1024 + h * HS + hl * 8;
    const unsigned short* k1p = k0p + (size_t)16 * HSTR;
    FragAB kb00, kb10, kb01, kb11;
    kb00.q[0] = *(const u32x4*)(k0p);      kb00.q[1] = *(const u32x4*)(k0p + 16);
    kb01.q[0] = *(const u32x4*)(k1p);      kb01.q[1] = *(const u32x4*)(k1p + 16);
    kb10.q[0] = *(const u32x4*)(k0p + 32); kb10.q[1] = *(const u32x4*)(k0p + 48);
    kb11.q[0] = *(const u32x4*)(k1p + 32); kb11.q[1] = *(const u32x4*)(k1p + 48);

    v8f S0 = {}, S1 = {};
    S0 = WMMA_BF16(qa0.v, kb00.v, S0);
    S0 = WMMA_BF16(qa1.v, kb10.v, S0);
    S1 = WMMA_BF16(qa0.v, kb01.v, S1);
    S1 = WMMA_BF16(qa1.v, kb11.v, S1);

    #pragma unroll
    for (int v = 0; v < 8; v++) {
      float s0 = S0[v] * 0.125f;    // 1/sqrt(64)
      float s1 = S1[v] * 0.125f;
      float tm = rowmax16(fmaxf(s0, s1));
      float mnew = fmaxf(mrow[v], tm);
      float c  = __expf(mrow[v] - mnew);
      float p0 = __expf(s0 - mnew);
      float p1 = __expf(s1 - mnew);
      mrow[v] = mnew;
      O0[v] *= c; O1[v] *= c; O2[v] *= c; O3[v] *= c; Lacc[v] *= c;
      int m = v + hl * 8;
      Pst[wave][m][r]      = f2bf(p0);
      Pst[wave][m][16 + r] = f2bf(p1);
    }

    FragAB pa;
    const unsigned short* pr = &Pst[wave][r][0];
    pa.q[0] = *(const u32x4*)(pr + hl * 8);
    pa.q[1] = *(const u32x4*)(pr + 16 + hl * 8);

    // Row sums: P @ ones -> replicated across the 16 columns.
    Lacc = WMMA_BF16(pa.v, ones.v, Lacc);

    const unsigned short* vp = vhead + (size_t)kt0 * HSTR + r;
    FragAB vb;
    #pragma unroll
    for (int s = 0; s < 16; s++) vb.u[s] = vp[(size_t)((s & 7) + hl * 8 + (s >> 3) * 16) * HSTR + 0];
    O0 = WMMA_BF16(pa.v, vb.v, O0);
    #pragma unroll
    for (int s = 0; s < 16; s++) vb.u[s] = vp[(size_t)((s & 7) + hl * 8 + (s >> 3) * 16) * HSTR + 16];
    O1 = WMMA_BF16(pa.v, vb.v, O1);
    #pragma unroll
    for (int s = 0; s < 16; s++) vb.u[s] = vp[(size_t)((s & 7) + hl * 8 + (s >> 3) * 16) * HSTR + 32];
    O2 = WMMA_BF16(pa.v, vb.v, O2);
    #pragma unroll
    for (int s = 0; s < 16; s++) vb.u[s] = vp[(size_t)((s & 7) + hl * 8 + (s >> 3) * 16) * HSTR + 48];
    O3 = WMMA_BF16(pa.v, vb.v, O3);
  }

  unsigned short* orow = o + (size_t)(b * T + t0) * 1024 + h * HS;
  #pragma unroll
  for (int v = 0; v < 8; v++) {
    float inv = 1.0f / Lacc[v];
    int m = v + hl * 8;
    unsigned short* op = orow + (size_t)m * 1024 + r;
    op[0]  = f2bf(O0[v] * inv);
    op[16] = f2bf(O1[v] * inv);
    op[32] = f2bf(O2[v] * inv);
    op[48] = f2bf(O3[v] * inv);
  }
}

// ---------------------------------------------------------------------------
// Host-side orchestration
// ---------------------------------------------------------------------------
extern "C" void kernel_launch(void* const* d_in, const int* in_sizes, int n_in,
                              void* d_out, int out_size, void* d_ws, size_t ws_size,
                              hipStream_t stream) {
  (void)in_sizes; (void)n_in; (void)out_size; (void)ws_size;
  const int Mrows = 8192;      // B*T
  const int D = 1024, N3 = 3072, FFN = 4096;

  const float* x      = (const float*)d_in[0];
  const float* ln1_g  = (const float*)d_in[1];
  const float* ln1_b  = (const float*)d_in[2];
  const float* W_ap   = (const float*)d_in[3];
  const float* b_ap   = (const float*)d_in[4];
  const float* W_qkv  = (const float*)d_in[5];
  const float* b_qkv  = (const float*)d_in[6];
  const float* W_proj = (const float*)d_in[7];
  const float* b_proj = (const float*)d_in[8];
  const float* ln2_g  = (const float*)d_in[9];
  const float* ln2_b  = (const float*)d_in[10];
  const float* W1     = (const float*)d_in[11];
  const float* b1     = (const float*)d_in[12];
  const float* W2     = (const float*)d_in[13];
  const float* b2     = (const float*)d_in[14];
  float* out = (float*)d_out;

  char* ws = (char*)d_ws;
  size_t off = 0;
  auto alloc = [&](size_t bytes) -> void* {
    void* p = ws + off;
    off += (bytes + 255) & ~(size_t)255;
    return p;
  };
  unsigned short* aln   = (unsigned short*)alloc((size_t)Mrows * D * 2);   // ln1; reused as f0
  unsigned short* abuf  = (unsigned short*)alloc((size_t)Mrows * D * 2);   // a; reused as o
  unsigned short* qkvb  = (unsigned short*)alloc((size_t)Mrows * N3 * 2);
  unsigned short* f1b   = (unsigned short*)alloc((size_t)Mrows * FFN * 2);
  float*          hbuf  = (float*)alloc((size_t)Mrows * D * 4);
  unsigned short* kTb   = (unsigned short*)alloc((size_t)Mrows * D * 2);   // [B*H][64][T]
  unsigned short* wapB  = (unsigned short*)alloc((size_t)D * D * 2);
  unsigned short* wqkvB = (unsigned short*)alloc((size_t)D * N3 * 2);
  unsigned short* wprjB = (unsigned short*)alloc((size_t)D * D * 2);
  unsigned short* w1B   = (unsigned short*)alloc((size_t)D * FFN * 2);
  unsigned short* w2B   = (unsigned short*)alloc((size_t)FFN * D * 2);
  unsigned short* f0    = aln;
  unsigned short* obuf  = abuf;

  // 1) weights fp32 -> bf16
  f2bf_kernel<<<2048, 256, 0, stream>>>(W_ap,  wapB,  D * D);
  f2bf_kernel<<<4096, 256, 0, stream>>>(W_qkv, wqkvB, D * N3);
  f2bf_kernel<<<2048, 256, 0, stream>>>(W_proj,wprjB, D * D);
  f2bf_kernel<<<4096, 256, 0, stream>>>(W1,    w1B,   D * FFN);
  f2bf_kernel<<<4096, 256, 0, stream>>>(W2,    w2B,   FFN * D);

  // 2) a_ln = LN(x)
  layernorm_kernel<false><<<Mrows, 256, 0, stream>>>(x, ln1_g, ln1_b, aln);

  // 3) a = a_ln @ W_ap + b_ap
  gemm_bf16_wmma<0, false, false, true><<<512, 256, 0, stream>>>(
      aln, wapB, b_ap, nullptr, nullptr, abuf, Mrows, D, D);

  // 4) qkv = a @ W_qkv + b_qkv
  gemm_bf16_wmma<0, false, false, true><<<1536, 256, 0, stream>>>(
      abuf, wqkvB, b_qkv, nullptr, nullptr, qkvb, Mrows, N3, D);

  // 5) kT = transpose(k) per head
  ktranspose_kernel<<<2048, 256, 0, stream>>>(qkvb, kTb);

  // 6) o = softmax(q k^T / 8) v
  attention_kernel<<<1024, 256, 0, stream>>>(qkvb, kTb, obuf);

  // 7) h = x + o @ W_proj + b_proj
  gemm_bf16_wmma<0, true, true, false><<<512, 256, 0, stream>>>(
      obuf, wprjB, b_proj, x, hbuf, nullptr, Mrows, D, D);

  // 8) f0 = gelu(LN(h))
  layernorm_kernel<true><<<Mrows, 256, 0, stream>>>(hbuf, ln2_g, ln2_b, f0);

  // 9) f1 = gelu(f0 @ W1 + b1)
  gemm_bf16_wmma<1, false, false, true><<<2048, 256, 0, stream>>>(
      f0, w1B, b1, nullptr, nullptr, f1b, Mrows, FFN, D);

  // 10) out = h + f1 @ W2 + b2
  gemm_bf16_wmma<0, true, true, false><<<512, 256, 0, stream>>>(
      f1b, w2B, b2, hbuf, out, nullptr, Mrows, D, FFN);
}
